// MultiHeadAttention_22660247453951
// MI455X (gfx1250) — compile-verified, matrix operand order checked by reading, not run on hardware
//
#include <hip/hip_runtime.h>

// ---------------------------------------------------------------------------
// MI455X (gfx1250) fused MHA, round 2.
//  - bf16 WMMA (v_wmma_f32_16x16x32_bf16) everywhere, f32 accumulate.
//  - One-shot f32->bf16 convert of X and W, then ALL GEMM tile staging is
//    GLOBAL_LOAD_ASYNC_TO_LDS_B128 (ASYNCcnt DMA, no VGPR round trip),
//    double-buffered in LDS.
//  - Attention: pass1 online max/sum, pass2 recompute scores, write normalized
//    weights once (268MB = traffic floor), attn = P@V via WMMA.
// Workspace: qv/kv/vt (24MB) + bf16 X (24MB) + bf16 W (6MB) = 54 MB.
// ---------------------------------------------------------------------------

typedef __attribute__((ext_vector_type(16))) __bf16 v16bf;
typedef __attribute__((ext_vector_type(8)))  float  v8f;

union Frag {
    unsigned int u[8];
    v16bf v;
};

__device__ __forceinline__ unsigned short f2bf(float f) {
    unsigned int u = __float_as_uint(f);
    u += 0x7FFFu + ((u >> 16) & 1u);   // round-to-nearest-even
    return (unsigned short)(u >> 16);
}

// 32-bit (bf16-pair) column index within a K=32 fragment row, per the CDNA5
// 16-bit A/B matrix VGPR layout (ISA 7.12.2).
__device__ __forceinline__ int kpair(int jj, int lane) {
    return (jj & 3) + ((jj >> 2) << 3) + (((lane >> 4) & 1) << 2);
}

// Async DMA: global -> LDS, 16B per lane, GVS addressing (saddr + 32-bit voff).
__device__ __forceinline__ void async_ld_b128(unsigned int lds_addr,
                                              unsigned int voff,
                                              unsigned long long sbase) {
    asm volatile("global_load_async_to_lds_b128 %0, %1, %2"
                 :: "v"(lds_addr), "v"(voff), "s"(sbase) : "memory");
}
__device__ __forceinline__ void wait_async0() {
    asm volatile("s_wait_asynccnt 0" ::: "memory");
}
__device__ __forceinline__ unsigned int lds_off(const void* p) {
    return (unsigned int)(unsigned long long)p;   // low 32b of generic = LDS offset
}

#define EDIM 1024

// ---------------------------------------------------------------------------
// Kernel 0: f32 -> bf16 (packed) convert, 4 elems/thread.
// ---------------------------------------------------------------------------
__global__ __launch_bounds__(256) void cvt_kernel(const float* __restrict__ src,
                                                  unsigned int* __restrict__ dst,
                                                  int n4) {
    int i = blockIdx.x * 256 + threadIdx.x;
    if (i < n4) {
        float4 x = ((const float4*)src)[i];
        unsigned int lo = (unsigned int)f2bf(x.x) | ((unsigned int)f2bf(x.y) << 16);
        unsigned int hi = (unsigned int)f2bf(x.z) | ((unsigned int)f2bf(x.w) << 16);
        ((uint2*)dst)[i] = make_uint2(lo, hi);
    }
}

// ---------------------------------------------------------------------------
// Kernel 1: C = Xbf @ Wbf^T + bias (f32 bias), bf16 out in view layout.
// vmode==0: out[bh][m][d] (Q,K)   vmode==1: out[bh][d][m] (V transposed)
// Block 256 (8 waves), tile 128x64, grid (16,32). Async double-buffered LDS.
// ---------------------------------------------------------------------------
__global__ __launch_bounds__(256) void proj_kernel(
    const unsigned short* __restrict__ Xb, const unsigned short* __restrict__ Wb,
    const float* __restrict__ bias, unsigned short* __restrict__ out, int vmode)
{
    // row = 64B data + 16B pad = 80B (20 uints): 16B-aligned for b128, and
    // bank stride 20 (gcd(20,64)=4, period 16) -> conflict-free frag reads.
    __shared__ __align__(16) unsigned int AsU[2][128 * 20];
    __shared__ __align__(16) unsigned int BsU[2][64 * 20];

    const int t    = threadIdx.x;
    const int lane = t & 31;
    const int w    = t >> 5;
    const int half = (lane >> 4) & 1;
    const int nidx = lane & 15;
    const int rowbase = blockIdx.y * 128;
    const int colbase = blockIdx.x * 64;

    const unsigned long long xb  = (unsigned long long)Xb;
    const unsigned long long wbb = (unsigned long long)Wb;
    unsigned int asb[2] = { lds_off(&AsU[0][0]), lds_off(&AsU[1][0]) };
    unsigned int bsb[2] = { lds_off(&BsU[0][0]), lds_off(&BsU[1][0]) };

    auto stage = [&](int buf, int k0) {
        // A tile: 128 rows x 64B   (2 async b128 per thread)
        #pragma unroll
        for (int i = 0; i < 2; ++i) {
            int idx = t + i * 256;
            int r = idx >> 2, c = idx & 3;
            async_ld_b128(asb[buf] + r * 80 + c * 16,
                          (unsigned int)((rowbase + r) * 2048 + k0 * 2 + c * 16), xb);
        }
        // B tile: 64 rows x 64B    (1 async b128 per thread)
        {
            int r = t >> 2, c = t & 3;
            async_ld_b128(bsb[buf] + r * 80 + c * 16,
                          (unsigned int)((colbase + r) * 2048 + k0 * 2 + c * 16), wbb);
        }
    };

    v8f acc[4] = {};
    stage(0, 0);
    for (int k0 = 0, it = 0; k0 < EDIM; k0 += 32, ++it) {
        const int buf = it & 1;
        wait_async0();
        __syncthreads();
        if (k0 + 32 < EDIM) stage(buf ^ 1, k0 + 32);

        const unsigned int* A = &AsU[buf][0];
        const unsigned int* B = &BsU[buf][0];
        Frag a;
        #pragma unroll
        for (int jj = 0; jj < 8; ++jj)
            a.u[jj] = A[(w * 16 + nidx) * 20 + kpair(jj, lane)];
        #pragma unroll
        for (int nt = 0; nt < 4; ++nt) {
            Frag b;
            #pragma unroll
            for (int jj = 0; jj < 8; ++jj)
                b.u[jj] = B[(nt * 16 + nidx) * 20 + kpair(jj, lane)];
            acc[nt] = __builtin_amdgcn_wmma_f32_16x16x32_bf16(
                false, a.v, false, b.v, (short)0, acc[nt], false, false);
        }
    }

    // epilogue: + bias, scatter into the scrambled view layout
    #pragma unroll
    for (int nt = 0; nt < 4; ++nt) {
        #pragma unroll
        for (int v = 0; v < 8; ++v) {
            int M = v + (half << 3);
            int r = rowbase + w * 16 + M;           // global row: b*N + n
            int e = colbase + nt * 16 + nidx;       // output feature
            float val = acc[nt][v] + bias[e];
            int b  = r >> 10;
            int n  = r & 1023;
            int bh = ((n & 3) << 4) + (e >> 6);
            int m  = (b << 8) + (n >> 2);
            int d  = e & 63;
            size_t idx = vmode ? ((size_t)bh * 64 + d) * 1024 + m
                               : ((size_t)bh * 1024 + m) * 64 + d;
            out[idx] = f2bf(val);
        }
    }
}

// ---------------------------------------------------------------------------
// Kernel 2: attention, one (bh, 128-row group) per block. 8 waves x 16 rows.
// grid (8, 64), block 256. Async double-buffered K/V tiles.
// ---------------------------------------------------------------------------
__global__ __launch_bounds__(256) void atten_kernel(
    const unsigned short* __restrict__ Qv, const unsigned short* __restrict__ Kv,
    const unsigned short* __restrict__ Vt, const float* __restrict__ bias,
    float* __restrict__ attn_out, float* __restrict__ w_out)
{
    // row = 128B data + 16B pad = 144B (36 uints): 16B aligned, bank stride 36
    // (gcd 4, 16-row period) -> conflict-free fragment reads.
    __shared__ __align__(16) unsigned int KsU[2][64 * 36];
    __shared__ __align__(16) unsigned int VsU[2][64 * 36];
    __shared__ __align__(16) unsigned short Ps[8 * 16 * 66];

    const int t    = threadIdx.x;
    const int lane = t & 31;
    const int w    = t >> 5;
    const int half = (lane >> 4) & 1;
    const int nidx = lane & 15;
    const int bh   = blockIdx.y;
    const int i0   = blockIdx.x * 128 + w * 16;

    const unsigned int* QvU = (const unsigned int*)Qv + (size_t)bh * 1024 * 32;
    const unsigned long long kvb = (unsigned long long)Kv + (size_t)bh * 131072;
    const unsigned long long vtb = (unsigned long long)Vt + (size_t)bh * 131072;
    const float* biasB = bias + (size_t)(bh & 3) * 1024 * 1024;

    unsigned int ksb[2] = { lds_off(&KsU[0][0]), lds_off(&KsU[1][0]) };
    unsigned int vsb[2] = { lds_off(&VsU[0][0]), lds_off(&VsU[1][0]) };

    auto stageK = [&](int buf, int j0) {   // 64 j-rows x 128B
        #pragma unroll
        for (int i = 0; i < 2; ++i) {
            int idx = t + i * 256;
            int r = idx >> 3, c = idx & 7;
            async_ld_b128(ksb[buf] + r * 144 + c * 16,
                          (unsigned int)((j0 + r) * 128 + c * 16), kvb);
        }
    };
    auto stageV = [&](int buf, int j0) {   // 64 d-rows x 128B (row stride 2KB)
        #pragma unroll
        for (int i = 0; i < 2; ++i) {
            int idx = t + i * 256;
            int r = idx >> 3, c = idx & 7;
            async_ld_b128(vsb[buf] + r * 144 + c * 16,
                          (unsigned int)(r * 2048 + j0 * 2 + c * 16), vtb);
        }
    };

    // Q tile (16 x 64) in registers: two K=32 A-fragments
    Frag qa[2];
    #pragma unroll
    for (int kh = 0; kh < 2; ++kh) {
        #pragma unroll
        for (int jj = 0; jj < 8; ++jj)
            qa[kh].u[jj] = QvU[(size_t)(i0 + nidx) * 32 + kh * 16 + kpair(jj, lane)];
    }

    // ----- pass 1: online row max / sum -----
    float runM[8], runS[8];
    #pragma unroll
    for (int v = 0; v < 8; ++v) { runM[v] = -3.0e38f; runS[v] = 0.0f; }

    stageK(0, 0);
    for (int j0 = 0, it = 0; j0 < 1024; j0 += 64, ++it) {
        const int buf = it & 1;
        wait_async0();
        __syncthreads();
        if (j0 + 64 < 1024) stageK(buf ^ 1, j0 + 64);

        const unsigned int* Ks = &KsU[buf][0];
        #pragma unroll
        for (int jn = 0; jn < 4; ++jn) {
            Frag b0, b1;
            #pragma unroll
            for (int jj = 0; jj < 8; ++jj) {
                b0.u[jj] = Ks[(jn * 16 + nidx) * 36 + kpair(jj, lane)];
                b1.u[jj] = Ks[(jn * 16 + nidx) * 36 + 16 + kpair(jj, lane)];
            }
            v8f s = {};
            s = __builtin_amdgcn_wmma_f32_16x16x32_bf16(false, qa[0].v, false, b0.v, (short)0, s, false, false);
            s = __builtin_amdgcn_wmma_f32_16x16x32_bf16(false, qa[1].v, false, b1.v, (short)0, s, false, false);
            int jcol = j0 + jn * 16 + nidx;
            #pragma unroll
            for (int v = 0; v < 8; ++v) {
                int row = i0 + v + (half << 3);
                float x = s[v] + biasB[(size_t)row * 1024 + jcol];
                float mn = fmaxf(runM[v], x);
                runS[v] = runS[v] * __expf(runM[v] - mn) + __expf(x - mn);
                runM[v] = mn;
            }
        }
    }
    // combine across the 16 lanes sharing each row (halves stay separate)
    #pragma unroll
    for (int v = 0; v < 8; ++v) {
        #pragma unroll
        for (int msk = 1; msk < 16; msk <<= 1) {
            float oM = __shfl_xor(runM[v], msk, 32);
            float oS = __shfl_xor(runS[v], msk, 32);
            float mn = fmaxf(runM[v], oM);
            runS[v] = runS[v] * __expf(runM[v] - mn) + oS * __expf(oM - mn);
            runM[v] = mn;
        }
    }
    float rinv[8];
    #pragma unroll
    for (int v = 0; v < 8; ++v) rinv[v] = 1.0f / runS[v];

    // ----- pass 2: normalized P -> weights out + attn = P @ V -----
    v8f aacc[4] = {};
    unsigned short* PsW = Ps + w * 1056;
    const unsigned int* PsU = (const unsigned int*)PsW;

    stageK(0, 0);
    stageV(0, 0);
    for (int j0 = 0, it = 0; j0 < 1024; j0 += 64, ++it) {
        const int buf = it & 1;
        wait_async0();
        __syncthreads();
        if (j0 + 64 < 1024) { stageK(buf ^ 1, j0 + 64); stageV(buf ^ 1, j0 + 64); }

        const unsigned int* Ks = &KsU[buf][0];
        const unsigned int* Vs = &VsU[buf][0];
        #pragma unroll
        for (int jn = 0; jn < 4; ++jn) {
            Frag b0, b1;
            #pragma unroll
            for (int jj = 0; jj < 8; ++jj) {
                b0.u[jj] = Ks[(jn * 16 + nidx) * 36 + kpair(jj, lane)];
                b1.u[jj] = Ks[(jn * 16 + nidx) * 36 + 16 + kpair(jj, lane)];
            }
            v8f s = {};
            s = __builtin_amdgcn_wmma_f32_16x16x32_bf16(false, qa[0].v, false, b0.v, (short)0, s, false, false);
            s = __builtin_amdgcn_wmma_f32_16x16x32_bf16(false, qa[1].v, false, b1.v, (short)0, s, false, false);
            int jcol = j0 + jn * 16 + nidx;
            #pragma unroll
            for (int v = 0; v < 8; ++v) {
                int M = v + (half << 3);
                int row = i0 + M;
                float x = s[v] + biasB[(size_t)row * 1024 + jcol];
                float p = __expf(x - runM[v]) * rinv[v];
                w_out[(size_t)bh * 1048576 + (size_t)row * 1024 + jcol] = p;
                PsW[M * 66 + jn * 16 + nidx] = f2bf(p);
            }
        }
        // wave-private C->A re-layout through LDS, then attn WMMAs
        #pragma unroll
        for (int jh = 0; jh < 2; ++jh) {
            Frag pa;
            #pragma unroll
            for (int jj = 0; jj < 8; ++jj)
                pa.u[jj] = PsU[nidx * 33 + jh * 16 + kpair(jj, lane)];
            #pragma unroll
            for (int dsb = 0; dsb < 4; ++dsb) {
                Frag vb;
                #pragma unroll
                for (int jj = 0; jj < 8; ++jj)
                    vb.u[jj] = Vs[(dsb * 16 + nidx) * 36 + jh * 16 + kpair(jj, lane)];
                aacc[dsb] = __builtin_amdgcn_wmma_f32_16x16x32_bf16(
                    false, pa.v, false, vb.v, (short)0, aacc[dsb], false, false);
            }
        }
    }

    // attn output: out[b=bh/16][n=i][e=(bh%16)*64+d]
    #pragma unroll
    for (int dsb = 0; dsb < 4; ++dsb) {
        #pragma unroll
        for (int v = 0; v < 8; ++v) {
            int M = v + (half << 3);
            int i = i0 + M;
            int d = dsb * 16 + nidx;
            attn_out[(size_t)(bh >> 4) * (1024u * 1024u) + (size_t)i * 1024 +
                     (bh & 15) * 64 + d] = aacc[dsb][v];
        }
    }
}

// ---------------------------------------------------------------------------
extern "C" void kernel_launch(void* const* d_in, const int* in_sizes, int n_in,
                              void* d_out, int out_size, void* d_ws, size_t ws_size,
                              hipStream_t stream) {
    (void)in_sizes; (void)n_in; (void)out_size; (void)ws_size;
    const float* query = (const float*)d_in[0];
    const float* key   = (const float*)d_in[1];
    const float* value = (const float*)d_in[2];
    const float* abias = (const float*)d_in[3];
    const float* Wq = (const float*)d_in[4];
    const float* bq = (const float*)d_in[5];
    const float* Wk = (const float*)d_in[6];
    const float* bk = (const float*)d_in[7];
    const float* Wv = (const float*)d_in[8];
    const float* bv = (const float*)d_in[9];

    unsigned short* ws = (unsigned short*)d_ws;
    const size_t M4 = (size_t)4 * 1024 * 1024;     // 4M elements
    const size_t M1 = (size_t)1024 * 1024;
    unsigned short* qv = ws;                        // [64][1024][64] bf16
    unsigned short* kv = qv + M4;                   // [64][1024][64]
    unsigned short* vt = kv + M4;                   // [64][64][1024]
    unsigned short* xq = vt + M4;                   // [4096][1024] bf16
    unsigned short* xk = xq + M4;
    unsigned short* xv = xk + M4;
    unsigned short* wq = xv + M4;                   // [1024][1024] bf16
    unsigned short* wk = wq + M1;
    unsigned short* wv = wk + M1;

    float* attn_out = (float*)d_out;                // [4][1024][1024]
    float* w_out    = attn_out + M4;                // [64][1024][1024]

    // f32 -> bf16 conversions
    cvt_kernel<<<dim3(4096), dim3(256), 0, stream>>>(query, (unsigned int*)xq, 1 << 20);
    cvt_kernel<<<dim3(4096), dim3(256), 0, stream>>>(key,   (unsigned int*)xk, 1 << 20);
    cvt_kernel<<<dim3(4096), dim3(256), 0, stream>>>(value, (unsigned int*)xv, 1 << 20);
    cvt_kernel<<<dim3(1024), dim3(256), 0, stream>>>(Wq,    (unsigned int*)wq, 1 << 18);
    cvt_kernel<<<dim3(1024), dim3(256), 0, stream>>>(Wk,    (unsigned int*)wk, 1 << 18);
    cvt_kernel<<<dim3(1024), dim3(256), 0, stream>>>(Wv,    (unsigned int*)wv, 1 << 18);

    dim3 gb(16, 32), tb(256);
    proj_kernel<<<gb, tb, 0, stream>>>(xq, wq, bq, qv, 0);
    proj_kernel<<<gb, tb, 0, stream>>>(xk, wk, bk, kv, 0);
    proj_kernel<<<gb, tb, 0, stream>>>(xv, wv, bv, vt, 1);

    dim3 ga(8, 64), ta(256);
    atten_kernel<<<ga, ta, 0, stream>>>(qv, kv, vt, abias, attn_out, w_out);
}
